// MoA_8761733284007
// MI455X (gfx1250) — matrix operand order repaired and checked by reading, not compiled, not run on hardware
//
#include <hip/hip_runtime.h>
#include <hip/hip_bf16.h>

typedef __attribute__((ext_vector_type(16))) __bf16 v16bf;
typedef __attribute__((ext_vector_type(8)))  float  v8f;

#define BM 128
#define BN 64
#define BKK 32
#define LDS_A 40   // bf16 elems per LDS row for 128x32 A tile (80B rows, 16B aligned)
#define LDS_B 40   // bf16 elems per LDS row for 64(n) x 32(k) Bt tile

// ---------------------------------------------------------------------------
// Generic WMMA GEMM: C[M,N] = act(A[M,K] @ W[K,N] + bias) + (R ? R : 0)
// act: 0=none, 1=relu, 2=gelu(tanh)
// 256 threads = 8 waves; wave w owns rows w*16..w*16+15 and all 4 col tiles.
// Software-pipelined: next K-tile global loads issue before current WMMAs.
// ---------------------------------------------------------------------------
__global__ __launch_bounds__(256)
void wmma_gemm(const float* __restrict__ A, const float* __restrict__ W,
               const float* __restrict__ bias, const float* __restrict__ Rres,
               float* __restrict__ C, int M, int N, int Kd, int act)
{
    __shared__ __bf16 As[BM * LDS_A];
    __shared__ __bf16 Bs[BN * LDS_B];

    const int tid  = threadIdx.x;
    const int lane = tid & 31;
    const int wave = tid >> 5;
    const int m0 = blockIdx.y * BM;
    const int n0 = blockIdx.x * BN;

    v8f acc[4] = {{}, {}, {}, {}};
    float4 ra[4];   // staged A regs: 128x32 = 1024 float4 / 256 thr = 4 each
    float4 rb[2];   // staged B regs: 32x64  =  512 float4 / 256 thr = 2 each

    auto load_tile = [&](int k0) {
#pragma unroll
        for (int i = 0; i < 4; ++i) {
            int idx = tid + i * 256;          // float4 index; 8 per row
            int row = idx >> 3;
            int col = (idx & 7) * 4;
            ra[i] = make_float4(0.f, 0.f, 0.f, 0.f);
            int gr = m0 + row;
            if (gr < M) ra[i] = *(const float4*)(A + (size_t)gr * Kd + k0 + col);
        }
#pragma unroll
        for (int i = 0; i < 2; ++i) {
            int idx = tid + i * 256;
            int kr = idx >> 4;                // 16 float4 per k-row
            int nc = (idx & 15) * 4;
            rb[i] = make_float4(0.f, 0.f, 0.f, 0.f);
            const float* src = W + (size_t)(k0 + kr) * N + n0 + nc;
            if (((N & 3) == 0) && (n0 + nc + 4 <= N)) {
                rb[i] = *(const float4*)src;
            } else {
                if (n0 + nc + 0 < N) rb[i].x = src[0];
                if (n0 + nc + 1 < N) rb[i].y = src[1];
                if (n0 + nc + 2 < N) rb[i].z = src[2];
                if (n0 + nc + 3 < N) rb[i].w = src[3];
            }
        }
    };

    auto publish_tile = [&]() {
#pragma unroll
        for (int i = 0; i < 4; ++i) {
            int idx = tid + i * 256;
            int row = idx >> 3;
            int col = (idx & 7) * 4;
            __bf16* dst = &As[row * LDS_A + col];
            dst[0] = (__bf16)ra[i].x; dst[1] = (__bf16)ra[i].y;
            dst[2] = (__bf16)ra[i].z; dst[3] = (__bf16)ra[i].w;
        }
#pragma unroll
        for (int i = 0; i < 2; ++i) {
            int idx = tid + i * 256;
            int kr = idx >> 4;
            int nc = (idx & 15) * 4;
            Bs[(nc + 0) * LDS_B + kr] = (__bf16)rb[i].x;
            Bs[(nc + 1) * LDS_B + kr] = (__bf16)rb[i].y;
            Bs[(nc + 2) * LDS_B + kr] = (__bf16)rb[i].z;
            Bs[(nc + 3) * LDS_B + kr] = (__bf16)rb[i].w;
        }
    };

    load_tile(0);                         // prologue: tile 0 in flight

    for (int k0 = 0; k0 < Kd; k0 += BKK) {
        __syncthreads();                  // prior compute done reading LDS
        publish_tile();                   // waits on loads, converts, ds_store
        __syncthreads();                  // LDS visible to all waves

        if (k0 + BKK < Kd) {
            load_tile(k0 + BKK);          // issue next global loads NOW
            __builtin_prefetch(A + (size_t)(m0 + ((tid & 31) << 2)) * Kd + k0 + 2 * BKK, 0, 1);
        }

        // ---- fragments + 4 WMMAs per wave, overlapped with loads above ----
        v16bf afrag;
        {
            int arow = wave * 16 + (lane & 15);
            int kg = lane >> 4;
            const __bf16* p = &As[arow * LDS_A + kg * 8];
            ((uint4*)&afrag)[0] = *(const uint4*)p;
            ((uint4*)&afrag)[1] = *(const uint4*)(p + 16);
        }
#pragma unroll
        for (int ct = 0; ct < 4; ++ct) {
            int ncol = ct * 16 + (lane & 15);
            int kg = lane >> 4;
            const __bf16* p = &Bs[ncol * LDS_B + kg * 8];
            v16bf bfrag;
            ((uint4*)&bfrag)[0] = *(const uint4*)p;
            ((uint4*)&bfrag)[1] = *(const uint4*)(p + 16);
            acc[ct] = __builtin_amdgcn_wmma_f32_16x16x32_bf16(
                false, afrag, false, bfrag, (short)0, acc[ct], false, false);
        }
    }

    // ---- epilogue: bias, activation, residual, bounded stores ----
#pragma unroll
    for (int ct = 0; ct < 4; ++ct) {
        int ncol = n0 + ct * 16 + (lane & 15);
        if (ncol < N) {
            float bv = bias ? bias[ncol] : 0.0f;
#pragma unroll
            for (int j = 0; j < 8; ++j) {
                int grow = m0 + wave * 16 + j + ((lane >> 4) << 3);
                if (grow < M) {
                    float v = acc[ct][j] + bv;
                    if (act == 1) v = fmaxf(v, 0.f);
                    else if (act == 2) {
                        float u = 0.7978845608028654f * (v + 0.044715f * v * v * v);
                        v = 0.5f * v * (1.0f + tanhf(u));
                    }
                    size_t off = (size_t)grow * N + ncol;
                    if (Rres) v += Rres[off];
                    C[off] = v;
                }
            }
        }
    }
}

// ---------------------------------------------------------------------------
// Fused flash attention, one block per (b,h,64-query tile). 4 waves x 16 rows.
// qkv: [B*N,1536] (q|k|v), out: [B*N,512]
// ---------------------------------------------------------------------------
#define ATS 72
__global__ __launch_bounds__(128)
void attn_kernel(const float* __restrict__ qkv, float* __restrict__ out,
                 int Bb, int Nn, float scale)
{
    __shared__ __bf16 Qs[64 * ATS];
    __shared__ __bf16 Ks[64 * ATS];
    __shared__ __bf16 Vs[64 * ATS];       // transposed: Vs[dh][key]
    __shared__ __bf16 Ps[4 * 16 * ATS];   // per-wave P staging

    const int tid  = threadIdx.x;
    const int lane = tid & 31;
    const int wave = tid >> 5;
    const int bh = blockIdx.y;
    const int b  = bh >> 3;
    const int hh = bh & 7;
    const int q0 = blockIdx.x * 64;
    const size_t LD = 1536;

    // ---- stage Q tile ----
    for (int idx = tid; idx < 64 * 16; idx += 128) {
        int r = idx >> 4, c4 = (idx & 15) * 4;
        float4 v = *(const float4*)(qkv + (size_t)(b * Nn + q0 + r) * LD + hh * 64 + c4);
        __bf16* d = &Qs[r * ATS + c4];
        d[0] = (__bf16)v.x; d[1] = (__bf16)v.y; d[2] = (__bf16)v.z; d[3] = (__bf16)v.w;
    }
    __syncthreads();

    v16bf aq0, aq1;
    {
        int r = wave * 16 + (lane & 15);
        int kg = lane >> 4;
        const __bf16* p = &Qs[r * ATS + kg * 8];
        ((uint4*)&aq0)[0] = *(const uint4*)p;
        ((uint4*)&aq0)[1] = *(const uint4*)(p + 16);
        const __bf16* p2 = &Qs[r * ATS + 32 + kg * 8];
        ((uint4*)&aq1)[0] = *(const uint4*)p2;
        ((uint4*)&aq1)[1] = *(const uint4*)(p2 + 16);
    }

    float m_i[8], l_i[8];
#pragma unroll
    for (int j = 0; j < 8; ++j) { m_i[j] = -1e30f; l_i[j] = 0.f; }
    v8f o0 = {}, o1 = {}, o2 = {}, o3 = {};

    for (int kt = 0; kt < Nn; kt += 64) {
        __syncthreads();
        // ---- stage K (row-major) and V (transposed) ----
        for (int idx = tid; idx < 64 * 16; idx += 128) {
            int r = idx >> 4, c4 = (idx & 15) * 4;
            const float* base = qkv + (size_t)(b * Nn + kt + r) * LD + hh * 64 + c4;
            float4 kv = *(const float4*)(base + 512);
            float4 vv = *(const float4*)(base + 1024);
            __bf16* dk = &Ks[r * ATS + c4];
            dk[0] = (__bf16)kv.x; dk[1] = (__bf16)kv.y;
            dk[2] = (__bf16)kv.z; dk[3] = (__bf16)kv.w;
            Vs[(c4 + 0) * ATS + r] = (__bf16)vv.x;
            Vs[(c4 + 1) * ATS + r] = (__bf16)vv.y;
            Vs[(c4 + 2) * ATS + r] = (__bf16)vv.z;
            Vs[(c4 + 3) * ATS + r] = (__bf16)vv.w;
        }
        __syncthreads();

        // ---- S = Q @ K^T (16x64 per wave) ----
        v8f s0 = {}, s1 = {}, s2 = {}, s3 = {};
#pragma unroll
        for (int ct = 0; ct < 4; ++ct) {
            v8f sc = {};
#pragma unroll
            for (int kc = 0; kc < 2; ++kc) {
                int key = ct * 16 + (lane & 15);
                int kh  = lane >> 4;
                const __bf16* p = &Ks[key * ATS + kc * 32 + kh * 8];
                v16bf bf;
                ((uint4*)&bf)[0] = *(const uint4*)p;
                ((uint4*)&bf)[1] = *(const uint4*)(p + 16);
                sc = __builtin_amdgcn_wmma_f32_16x16x32_bf16(
                    false, kc ? aq1 : aq0, false, bf, (short)0, sc, false, false);
            }
            if (ct == 0) s0 = sc; else if (ct == 1) s1 = sc;
            else if (ct == 2) s2 = sc; else s3 = sc;
        }

        // ---- online softmax (row stats via 16-lane shuffles) ----
#pragma unroll
        for (int j = 0; j < 8; ++j) {
            float v0 = s0[j] * scale, v1 = s1[j] * scale;
            float v2 = s2[j] * scale, v3 = s3[j] * scale;
            float mx = fmaxf(fmaxf(v0, v1), fmaxf(v2, v3));
#pragma unroll
            for (int msk = 1; msk < 16; msk <<= 1) mx = fmaxf(mx, __shfl_xor(mx, msk, 32));
            float mnew = fmaxf(m_i[j], mx);
            float p0 = __expf(v0 - mnew), p1 = __expf(v1 - mnew);
            float p2 = __expf(v2 - mnew), p3 = __expf(v3 - mnew);
            float rs = p0 + p1 + p2 + p3;
#pragma unroll
            for (int msk = 1; msk < 16; msk <<= 1) rs += __shfl_xor(rs, msk, 32);
            float alpha = __expf(m_i[j] - mnew);
            l_i[j] = l_i[j] * alpha + rs;
            m_i[j] = mnew;
            o0[j] *= alpha; o1[j] *= alpha; o2[j] *= alpha; o3[j] *= alpha;
            s0[j] = p0; s1[j] = p1; s2[j] = p2; s3[j] = p3;
        }

        // ---- relayout P: C-layout -> A fragments via per-wave LDS ----
        __bf16* pw = &Ps[wave * 16 * ATS];
        {
            int colb = lane & 15;
            int rofs = (lane >> 4) * 8;
#pragma unroll
            for (int j = 0; j < 8; ++j) {
                pw[(rofs + j) * ATS +  0 + colb] = (__bf16)s0[j];
                pw[(rofs + j) * ATS + 16 + colb] = (__bf16)s1[j];
                pw[(rofs + j) * ATS + 32 + colb] = (__bf16)s2[j];
                pw[(rofs + j) * ATS + 48 + colb] = (__bf16)s3[j];
            }
        }
        v16bf ap0, ap1;
        {
            int r = lane & 15, kg = lane >> 4;
            const __bf16* p = &pw[r * ATS + kg * 8];
            ((uint4*)&ap0)[0] = *(const uint4*)p;
            ((uint4*)&ap0)[1] = *(const uint4*)(p + 16);
            const __bf16* p2 = &pw[r * ATS + 32 + kg * 8];
            ((uint4*)&ap1)[0] = *(const uint4*)p2;
            ((uint4*)&ap1)[1] = *(const uint4*)(p2 + 16);
        }

        // ---- O += P @ V ----
#pragma unroll
        for (int ct = 0; ct < 4; ++ct) {
            v8f oc = (ct == 0) ? o0 : (ct == 1) ? o1 : (ct == 2) ? o2 : o3;
#pragma unroll
            for (int kc = 0; kc < 2; ++kc) {
                int dh = ct * 16 + (lane & 15);
                int kh = lane >> 4;
                const __bf16* p = &Vs[dh * ATS + kc * 32 + kh * 8];
                v16bf bf;
                ((uint4*)&bf)[0] = *(const uint4*)p;
                ((uint4*)&bf)[1] = *(const uint4*)(p + 16);
                oc = __builtin_amdgcn_wmma_f32_16x16x32_bf16(
                    false, kc ? ap1 : ap0, false, bf, (short)0, oc, false, false);
            }
            if (ct == 0) o0 = oc; else if (ct == 1) o1 = oc;
            else if (ct == 2) o2 = oc; else o3 = oc;
        }
    }

    // ---- normalize + store ----
    int colb = lane & 15;
    int rofs = (lane >> 4) * 8;
#pragma unroll
    for (int j = 0; j < 8; ++j) {
        float inv = 1.0f / l_i[j];
        int grow = b * Nn + q0 + wave * 16 + rofs + j;
        float* dst = out + (size_t)grow * 512 + hh * 64;
        dst[ 0 + colb] = o0[j] * inv;
        dst[16 + colb] = o1[j] * inv;
        dst[32 + colb] = o2[j] * inv;
        dst[48 + colb] = o3[j] * inv;
    }
}

// ---------------------------------------------------------------------------
// LayerNorm over rows of length D
// ---------------------------------------------------------------------------
__global__ __launch_bounds__(256)
void layernorm_rows(const float* __restrict__ X, const float* __restrict__ gg,
                    const float* __restrict__ bb, float* __restrict__ Y, int D)
{
    __shared__ float sred[16];
    const int row = blockIdx.x;
    const float* x = X + (size_t)row * D;
    float* y = Y + (size_t)row * D;
    const int tid = threadIdx.x, lane = tid & 31, wave = tid >> 5;
    const int nw = blockDim.x >> 5;

    float s = 0.f;
    for (int i = tid; i < D; i += blockDim.x) s += x[i];
    for (int m = 16; m >= 1; m >>= 1) s += __shfl_xor(s, m, 32);
    if (lane == 0) sred[wave] = s;
    __syncthreads();
    if (tid == 0) { float t = 0; for (int w = 0; w < nw; ++w) t += sred[w]; sred[0] = t / D; }
    __syncthreads();
    float mean = sred[0];

    float v = 0.f;
    for (int i = tid; i < D; i += blockDim.x) { float d = x[i] - mean; v += d * d; }
    for (int m = 16; m >= 1; m >>= 1) v += __shfl_xor(v, m, 32);
    if (lane == 0) sred[8 + wave] = v;
    __syncthreads();
    if (tid == 0) { float t = 0; for (int w = 0; w < nw; ++w) t += sred[8 + w]; sred[1] = rsqrtf(t / D + 1e-5f); }
    __syncthreads();
    float rstd = sred[1];
    for (int i = tid; i < D; i += blockDim.x)
        y[i] = (x[i] - mean) * rstd * gg[i] + bb[i];
}

// ---------------------------------------------------------------------------
// Y[b,d] = mean_n X[b,n,d]
// ---------------------------------------------------------------------------
__global__ void mean_over_n(const float* __restrict__ X, float* __restrict__ Y,
                            int Bb, int Nn, int D)
{
    int idx = blockIdx.x * blockDim.x + threadIdx.x;
    if (idx >= Bb * D) return;
    int b = idx / D, d = idx % D;
    const float* p = X + (size_t)b * Nn * D + d;
    float s = 0.f;
    for (int n = 0; n < Nn; ++n) s += p[(size_t)n * D];
    Y[idx] = s / (float)Nn;
}

// ---------------------------------------------------------------------------
// softmax over K=4, top-2 normalized combine weights
// ---------------------------------------------------------------------------
__global__ void router_finalize(const float* __restrict__ rl,
                                float* __restrict__ gsoft_out,
                                float* __restrict__ Wfull)
{
    int b = threadIdx.x;
    if (b >= 16) return;
    float v[4];
    float mx = -1e30f;
    for (int k = 0; k < 4; ++k) { v[k] = rl[b * 4 + k]; mx = fmaxf(mx, v[k]); }
    float s = 0.f;
    for (int k = 0; k < 4; ++k) { v[k] = __expf(v[k] - mx); s += v[k]; }
    for (int k = 0; k < 4; ++k) v[k] /= s;
    int i1 = 0;
    for (int k = 1; k < 4; ++k) if (v[k] > v[i1]) i1 = k;
    int i2 = -1;
    for (int k = 0; k < 4; ++k) { if (k == i1) continue; if (i2 < 0 || v[k] > v[i2]) i2 = k; }
    float wsum = v[i1] + v[i2] + 1e-8f;
    for (int k = 0; k < 4; ++k) {
        float w = (k == i1) ? v[i1] / wsum : (k == i2) ? v[i2] / wsum : 0.f;
        Wfull[b * 4 + k] = w;
        gsoft_out[b * 4 + k] = v[k];
    }
}

// ---------------------------------------------------------------------------
// z[b,d] = sum_e Wfull[b,e] * latents[e,b,d]
// ---------------------------------------------------------------------------
__global__ void combine_latents(const float* __restrict__ lat,
                                const float* __restrict__ Wfull,
                                float* __restrict__ z, int Bb, int D, int Kk)
{
    int idx = blockIdx.x * blockDim.x + threadIdx.x;
    if (idx >= Bb * D) return;
    int b = idx / D, d = idx % D;
    float s = 0.f;
    for (int e = 0; e < Kk; ++e)
        s += Wfull[b * Kk + e] * lat[((size_t)e * Bb + b) * D + d];
    z[idx] = s;
}

// ---------------------------------------------------------------------------
extern "C" void kernel_launch(void* const* d_in, const int* in_sizes, int n_in,
                              void* d_out, int out_size, void* d_ws, size_t ws_size,
                              hipStream_t stream)
{
    const float* x      = (const float*)d_in[0];
    const float* proj_w = (const float*)d_in[1];
    const float* proj_b = (const float*)d_in[2];
    const float* qkv_w  = (const float*)d_in[3];
    const float* ao_w   = (const float*)d_in[4];
    const float* ao_b   = (const float*)d_in[5];
    const float* ln1_g  = (const float*)d_in[6];
    const float* ln1_b  = (const float*)d_in[7];
    const float* ln2_g  = (const float*)d_in[8];
    const float* ln2_b  = (const float*)d_in[9];
    const float* mlp_w1 = (const float*)d_in[10];
    const float* mlp_b1 = (const float*)d_in[11];
    const float* mlp_w2 = (const float*)d_in[12];
    const float* mlp_b2 = (const float*)d_in[13];
    const float* lnf_g  = (const float*)d_in[14];
    const float* lnf_b  = (const float*)d_in[15];
    const float* rw1 = (const float*)d_in[16];
    const float* rb1 = (const float*)d_in[17];
    const float* rw2 = (const float*)d_in[18];
    const float* rb2 = (const float*)d_in[19];
    const float* hw1 = (const float*)d_in[20];
    const float* hb1 = (const float*)d_in[21];
    const float* hw2 = (const float*)d_in[22];
    const float* hb2 = (const float*)d_in[23];

    float* out = (float*)d_out;
    float* ws  = (float*)d_ws;

    constexpr int Bc = 16, Nc = 512, IN = 2048, D = 512, Kx = 4, DEP = 2,
                  HD = 8, QKV3 = 1536, MLPD = 512;
    const int BNr = Bc * Nc; // 8192 rows

    size_t o = 0;
    float* Hb    = ws + o; o += (size_t)BNr * D;
    float* hb    = ws + o; o += (size_t)BNr * D;
    float* yb    = ws + o; o += (size_t)BNr * D;
    float* qkvb  = ws + o; o += (size_t)BNr * QKV3;
    float* aob   = ws + o; o += (size_t)BNr * D;
    float* tb    = ws + o; o += (size_t)BNr * D;
    float* bag   = ws + o; o += (size_t)Bc * D;
    float* r1    = ws + o; o += (size_t)Bc * 256;
    float* rlb   = ws + o; o += (size_t)Bc * Kx;
    float* Wfull = ws + o; o += (size_t)Bc * Kx;
    float* lat   = ws + o; o += (size_t)Kx * Bc * D;
    float* hhb   = ws + o; o += (size_t)Bc * 128;

    dim3 blk(256);
    auto gg = [](int M, int N) { return dim3((N + 63) / 64, (M + 127) / 128); };

    // shared projection
    wmma_gemm<<<gg(BNr, D), blk, 0, stream>>>(x, proj_w, proj_b, nullptr, Hb, BNr, D, IN, 0);
    // router
    mean_over_n<<<(Bc * D + 255) / 256, 256, 0, stream>>>(Hb, bag, Bc, Nc, D);
    wmma_gemm<<<gg(Bc, 256), blk, 0, stream>>>(bag, rw1, rb1, nullptr, r1, Bc, 256, D, 1);
    wmma_gemm<<<gg(Bc, Kx), blk, 0, stream>>>(r1, rw2, rb2, nullptr, rlb, Bc, Kx, 256, 0);
    router_finalize<<<1, 32, 0, stream>>>(rlb, out + 8192 + 160, Wfull);

    // experts (sequential, reuse buffers)
    for (int e = 0; e < Kx; ++e) {
        hipMemcpyAsync(hb, Hb, (size_t)BNr * D * sizeof(float),
                       hipMemcpyDeviceToDevice, stream);
        for (int l = 0; l < DEP; ++l) {
            int el = e * DEP + l;
            layernorm_rows<<<BNr, 256, 0, stream>>>(hb, ln1_g + (size_t)el * D,
                                                    ln1_b + (size_t)el * D, yb, D);
            wmma_gemm<<<gg(BNr, QKV3), blk, 0, stream>>>(
                yb, qkv_w + (size_t)el * D * QKV3, nullptr, nullptr, qkvb, BNr, QKV3, D, 0);
            attn_kernel<<<dim3(Nc / 64, Bc * HD), 128, 0, stream>>>(qkvb, aob, Bc, Nc, 0.125f);
            wmma_gemm<<<gg(BNr, D), blk, 0, stream>>>(
                aob, ao_w + (size_t)el * D * D, ao_b + (size_t)el * D, hb, hb, BNr, D, D, 0);
            layernorm_rows<<<BNr, 256, 0, stream>>>(hb, ln2_g + (size_t)el * D,
                                                    ln2_b + (size_t)el * D, yb, D);
            wmma_gemm<<<gg(BNr, MLPD), blk, 0, stream>>>(
                yb, mlp_w1 + (size_t)el * D * MLPD, mlp_b1 + (size_t)el * MLPD,
                nullptr, tb, BNr, MLPD, D, 2);
            wmma_gemm<<<gg(BNr, D), blk, 0, stream>>>(
                tb, mlp_w2 + (size_t)el * MLPD * D, mlp_b2 + (size_t)el * D, hb, hb, BNr, D, MLPD, 0);
        }
        layernorm_rows<<<BNr, 256, 0, stream>>>(hb, lnf_g + (size_t)e * D,
                                                lnf_b + (size_t)e * D, yb, D);
        mean_over_n<<<(Bc * D + 255) / 256, 256, 0, stream>>>(yb, lat + (size_t)e * Bc * D,
                                                              Bc, Nc, D);
    }

    // combine + head
    combine_latents<<<(Bc * D + 255) / 256, 256, 0, stream>>>(lat, Wfull, out, Bc, D, Kx);
    wmma_gemm<<<gg(Bc, 128), blk, 0, stream>>>(out, hw1, hb1, nullptr, hhb, Bc, 128, D, 1);
    wmma_gemm<<<gg(Bc, 10), blk, 0, stream>>>(hhb, hw2, hb2, nullptr, out + 8192, Bc, 10, 128, 0);

    (void)in_sizes; (void)n_in; (void)out_size; (void)ws_size;
}
